// NSDEModel_19645180412411
// MI455X (gfx1250) — matrix operand order repaired and checked by reading, not compiled, not run on hardware
//
#include <hip/hip_runtime.h>
#include <hip/hip_bf16.h>
#include <math.h>

typedef _Float16 v16h  __attribute__((ext_vector_type(16)));
typedef float    v8f   __attribute__((ext_vector_type(8)));
typedef unsigned int u32x4 __attribute__((ext_vector_type(4)));
typedef int      i32x4 __attribute__((ext_vector_type(4)));
typedef int      i32x8 __attribute__((ext_vector_type(8)));

#define kB     16384
#define kT     128
#define kF     64
#define kH     64
#define kSteps 20

// Swizzled-weight offsets (in _Float16 units). Each matrix is stored as packed
// WMMA B-fragments: fragment (kc, nt) occupies 512 halves; lane L holds 16
// contiguous halves (32 bytes) = its v16h operand.
#define OFF_W1 0        // dW1: 128x128 -> 4 kc x 8 nt
#define OFF_W2 16384    // dW2: 128x128
#define OFF_W3 32768    // dW3: 128x64  -> 4 kc x 4 nt
#define OFF_G1 40960    // gW1: 64x64   -> 2 kc x 4 nt
#define OFF_G2 45056    // gW2: 64x64
#define OFF_R1 49152    // rW1: 64x32   -> 2 kc x 2 nt
#define OFF_R2 51200    // rW2: 32x16 (zero-padded from 32x2) -> 1 frag
#define WS_HALVES 51712           // 103,424 bytes
#define ACT_HALVES (4 * 4096)     // 4 waves x 2x(16x128) f16 ping-pong buffers
#define LDS_HALVES (ACT_HALVES + WS_HALVES)
#define LDS_BYTES  (LDS_HALVES * 2)   // 136,192 B (< 320 KB WGP LDS)

// ---------------------------------------------------------------------------
// Prep: scatter f32 weights into f16 B-fragment layout in d_ws.
// B layout (16-bit, 32x16): lanes 0-15 hold K=0..7,16..23 (N = lane), lanes 16-31
// hold K=8..15,24..31 (N = lane-16); element = (K&7) + (K&16 ? 8 : 0).
// ---------------------------------------------------------------------------
__global__ void prep_weights(const float* __restrict__ dW1, const float* __restrict__ dW2,
                             const float* __restrict__ dW3, const float* __restrict__ gW1,
                             const float* __restrict__ gW2, const float* __restrict__ rW1,
                             const float* __restrict__ rW2, _Float16* __restrict__ ws) {
  int gid = blockIdx.x * blockDim.x + threadIdx.x;
  if (gid >= WS_HALVES) return;
  int start, Ntot; const float* src; bool pad = false;
  if      (gid < OFF_W2) { start = OFF_W1; Ntot = 128; src = dW1; }
  else if (gid < OFF_W3) { start = OFF_W2; Ntot = 128; src = dW2; }
  else if (gid < OFF_G1) { start = OFF_W3; Ntot = 64;  src = dW3; }
  else if (gid < OFF_G2) { start = OFF_G1; Ntot = 64;  src = gW1; }
  else if (gid < OFF_R1) { start = OFF_G2; Ntot = 64;  src = gW2; }
  else if (gid < OFF_R2) { start = OFF_R1; Ntot = 32;  src = rW1; }
  else                   { start = OFF_R2; Ntot = 16;  src = rW2; pad = true; }
  int e = gid - start;
  int k = e / Ntot, n = e % Ntot;
  float v = pad ? ((n < 2) ? rW2[k * 2 + n] : 0.0f) : src[e];
  int kc = k >> 5, r = k & 31, nt = n >> 4, nn = n & 15;
  int ntTot = Ntot >> 4;
  int lane = (((r >> 3) & 1) ? 16 : 0) + nn;
  int el = (r & 7) + ((r & 16) ? 8 : 0);
  ws[start + ((kc * ntTot + nt) * 32 + lane) * 16 + el] = (_Float16)v;
}

// ---------------------------------------------------------------------------
// WMMA helpers
// ---------------------------------------------------------------------------
__device__ __forceinline__ v8f wmma_f16(v16h a, v16h b, v8f c) {
  return __builtin_amdgcn_wmma_f32_16x16x32_f16(false, a, false, b, (short)0, c,
                                                false, false);
}

// A-fragment (16x32 f16) address for activation element (M, K), per ISA 7.12.2.
__device__ __forceinline__ int a_addr(int M, int K) {
  int kc = K >> 5, r = K & 31;
  int lg = (r >> 3) & 1;
  int el = (r & 7) + ((r & 16) ? 8 : 0);
  return (kc * 32 + lg * 16 + M) * 16 + el;
}

// Scatter a C fragment (f32) into the A layout (f16) of the next layer's input.
__device__ __forceinline__ void store_c_as_a(_Float16* buf, v8f c, int ntile, int lane) {
  int nn = lane & 15, grp = lane >> 4;
  int K = ntile * 16 + nn;
  int kc = K >> 5, r = K & 31;
  int lg = (r >> 3) & 1;
  int el = (r & 7) + ((r & 16) ? 8 : 0);
  _Float16* p = buf + (kc * 32 + lg * 16) * 16 + el;
#pragma unroll
  for (int v = 0; v < 8; ++v) p[(grp * 8 + v) * 16] = (_Float16)c[v];
}

// One 16x16 output tile: sum over KC kchunks, A and B both LDS-resident.
template <int KC>
__device__ __forceinline__ v8f gemm_tile(const _Float16* aBuf, const _Float16* wFrag,
                                         int ntile, int ntTot, int lane) {
  v8f acc = {};
#pragma unroll
  for (int kc = 0; kc < KC; ++kc) {
    v16h a = *(const v16h*)(aBuf + (kc * 32 + lane) * 16);
    v16h b = *(const v16h*)(wFrag + ((kc * ntTot + ntile) * 32 + lane) * 16);
    acc = wmma_f16(a, b, acc);
  }
  return acc;
}

// TDM: DMA the whole swizzled weight blob (103,424 B = 12,928 x 8B elements,
// 1-D tile) from global d_ws into LDS at byte offset ldsOff. D# built per
// ISA ch.8: group0 = {count|flags, lds_addr, ga_lo, ga_hi|type=2},
// group1 packs data_size=8B, tensor_dim0/tile_dim0 = 12928.
__device__ __forceinline__ void tdm_load_weights(const _Float16* ws, unsigned ldsOff) {
  unsigned long long ga = (unsigned long long)(size_t)ws;
  const unsigned NE = WS_HALVES / 4;  // 12,928 8-byte elements
  u32x4 g0 = {1u,                                   // count=1, user mode
              ldsOff,                               // lds_addr
              (unsigned)(ga & 0xffffffffu),         // global_addr[31:0]
              (unsigned)((ga >> 32) & 0x01ffffffu) | (2u << 30)};  // ga[56:32] | type=2
  i32x8 g1 = {(int)(3u << 16),                      // data_size=3 (8B), mask=0
              (int)((NE & 0xffffu) << 16),          // tensor_dim0[15:0]
              (int)((NE >> 16) | (1u << 16)),       // tensor_dim0[31:16] | tensor_dim1=1
              (int)((NE & 0xffffu) << 16),          // tile_dim0 = NE (1-D tile)
              0,                                    // tile_dim1=0, tile_dim2=0
              (int)NE,                              // tensor_dim0_stride[31:0]
              0, 0};
  i32x4 gz = {0, 0, 0, 0};
#if defined(__clang_major__) && (__clang_major__ >= 23)
  i32x8 gz8 = {0, 0, 0, 0, 0, 0, 0, 0};
  __builtin_amdgcn_tensor_load_to_lds(g0, g1, gz, gz, gz8, 0);
#else
  __builtin_amdgcn_tensor_load_to_lds(g0, g1, gz, gz, 0);
#endif
}

// ---------------------------------------------------------------------------
// Main kernel: 4 waves/block, each wave integrates 16 SDE paths.
// Dynamic LDS: [0, 32KB) activation ping-pong buffers, [32KB, 133KB) weights.
// ---------------------------------------------------------------------------
__global__ void __launch_bounds__(128) nsde_kernel(
    const float* __restrict__ x_path, const float* __restrict__ t_span,
    const float* __restrict__ noise, const _Float16* __restrict__ ws,
    const float* __restrict__ db1, const float* __restrict__ db2,
    const float* __restrict__ db3, const float* __restrict__ gb1,
    const float* __restrict__ gb2, const float* __restrict__ gscale,
    const float* __restrict__ rb1, const float* __restrict__ rb2,
    float* __restrict__ out) {
  extern __shared__ _Float16 smem[];
  const int tid = threadIdx.x;
  const int w = tid >> 5, lane = tid & 31;
  const int nn = lane & 15, grp = lane >> 4;
  const int b0 = blockIdx.x * 64 + w * 16;
  _Float16* bufA = smem + w * 4096;          // layer input  (A layout, K<=128)
  _Float16* bufB = bufA + 2048;              // layer output (A layout)
  _Float16* wlds = smem + ACT_HALVES;        // LDS-resident swizzled weights

  // One TDM descriptor per block pulls all weights into LDS.
  if (w == 0) {
    unsigned ldsOff = (unsigned)(size_t)(void*)wlds;  // LDS aperture: low 32 = offset
    tdm_load_weights(ws, ldsOff);
    __builtin_amdgcn_s_wait_tensorcnt(0);
  }
  __syncthreads();

  v8f h[4];                                  // h state: 16x64 as 4 C fragments
#pragma unroll
  for (int nt = 0; nt < 4; ++nt) h[nt] = (v8f){};

  const float tmax = t_span[kSteps];

#pragma unroll 1
  for (int s = 0; s < kSteps; ++s) {
    float t = t_span[s];
    float dt = t_span[s + 1] - t;
    float sq = sqrtf(dt);
    int idx = (int)(t / tmax * (kT - 1));
    idx = idx < 0 ? 0 : (idx > kT - 1 ? kT - 1 : idx);

    // Stage drift input z = [h | x_t] into bufA (A layout). h -> K=0..63.
#pragma unroll
    for (int nt = 0; nt < 4; ++nt) store_c_as_a(bufA, h[nt], nt, lane);
    // x_t -> K=64..127 (coalesced f32 reads of the selected time slice).
#pragma unroll
    for (int j = 0; j < 32; ++j) {
      int i = j * 32 + lane;                 // 16 rows x 64 features
      int row = i >> 6, f = i & 63;
      float xv = x_path[(((size_t)(b0 + row)) * kT + idx) * kF + f];
      bufA[a_addr(row, 64 + f)] = (_Float16)xv;
    }

    // Diffusion L1: relu(h @ gW1 + gb1) -> bufB (K=0..63)
#pragma unroll
    for (int nt = 0; nt < 4; ++nt) {
      v8f a = gemm_tile<2>(bufA, wlds + OFF_G1, nt, 4, lane);
      float bb = gb1[nt * 16 + nn];
      v8f r;
#pragma unroll
      for (int v = 0; v < 8; ++v) { float x = a[v] + bb; r[v] = x > 0.f ? x : 0.f; }
      store_c_as_a(bufB, r, nt, lane);
    }
    // Diffusion L2: g = gscale * sigmoid(. @ gW2 + gb2), held in registers.
    v8f g[4];
#pragma unroll
    for (int nt = 0; nt < 4; ++nt) {
      v8f a = gemm_tile<2>(bufB, wlds + OFF_G2, nt, 4, lane);
      float bb = gb2[nt * 16 + nn];
      float sc = gscale[nt * 16 + nn];
#pragma unroll
      for (int v = 0; v < 8; ++v) {
        float x = a[v] + bb;
        g[nt][v] = sc / (1.0f + __expf(-x));
      }
    }
    // Drift L1: relu([h|x] @ dW1 + db1) -> bufB (K=0..127)
#pragma unroll
    for (int nt = 0; nt < 8; ++nt) {
      v8f a = gemm_tile<4>(bufA, wlds + OFF_W1, nt, 8, lane);
      float bb = db1[nt * 16 + nn];
      v8f r;
#pragma unroll
      for (int v = 0; v < 8; ++v) { float x = a[v] + bb; r[v] = x > 0.f ? x : 0.f; }
      store_c_as_a(bufB, r, nt, lane);
    }
    // Drift L2: relu(. @ dW2 + db2) -> bufA (overwrite; [h|x] consumed)
#pragma unroll
    for (int nt = 0; nt < 8; ++nt) {
      v8f a = gemm_tile<4>(bufB, wlds + OFF_W2, nt, 8, lane);
      float bb = db2[nt * 16 + nn];
      v8f r;
#pragma unroll
      for (int v = 0; v < 8; ++v) { float x = a[v] + bb; r[v] = x > 0.f ? x : 0.f; }
      store_c_as_a(bufA, r, nt, lane);
    }
    // Drift L3 + Euler-Maruyama update (noise loaded in C layout).
    const float* zs = noise + (size_t)s * kB * kH;
#pragma unroll
    for (int nt = 0; nt < 4; ++nt) {
      v8f a = gemm_tile<4>(bufA, wlds + OFF_W3, nt, 4, lane);
      float bb = db3[nt * 16 + nn];
#pragma unroll
      for (int v = 0; v < 8; ++v) {
        int bi = b0 + grp * 8 + v;
        float z = zs[(size_t)bi * kH + nt * 16 + nn];
        h[nt][v] += (a[v] + bb) * dt + g[nt][v] * sq * z;
      }
    }
  }

  // Readout: relu(h @ rW1 + rb1) @ rW2 + rb2
#pragma unroll
  for (int nt = 0; nt < 4; ++nt) store_c_as_a(bufA, h[nt], nt, lane);
#pragma unroll
  for (int nt = 0; nt < 2; ++nt) {
    v8f a = gemm_tile<2>(bufA, wlds + OFF_R1, nt, 2, lane);
    float bb = rb1[nt * 16 + nn];
    v8f r;
#pragma unroll
    for (int v = 0; v < 8; ++v) { float x = a[v] + bb; r[v] = x > 0.f ? x : 0.f; }
    store_c_as_a(bufB, r, nt, lane);
  }
  {
    v8f a = gemm_tile<1>(bufB, wlds + OFF_R2, 0, 1, lane);
    if (nn < 2) {                      // only cols 0 (mu) and 1 (log_sigma) valid
      float bb = rb2[nn];
#pragma unroll
      for (int v = 0; v < 8; ++v) {
        int bi = b0 + grp * 8 + v;
        out[(size_t)nn * kB + bi] = a[v] + bb;
      }
    }
  }
}

// ---------------------------------------------------------------------------
extern "C" void kernel_launch(void* const* d_in, const int* in_sizes, int n_in,
                              void* d_out, int out_size, void* d_ws, size_t ws_size,
                              hipStream_t stream) {
  const float* x_path = (const float*)d_in[0];
  const float* t_span = (const float*)d_in[1];
  const float* dWn    = (const float*)d_in[2];
  const float* dW1 = (const float*)d_in[3];
  const float* db1 = (const float*)d_in[4];
  const float* dW2 = (const float*)d_in[5];
  const float* db2 = (const float*)d_in[6];
  const float* dW3 = (const float*)d_in[7];
  const float* db3 = (const float*)d_in[8];
  const float* gW1 = (const float*)d_in[9];
  const float* gb1 = (const float*)d_in[10];
  const float* gW2 = (const float*)d_in[11];
  const float* gb2 = (const float*)d_in[12];
  const float* gsc = (const float*)d_in[13];
  const float* rW1 = (const float*)d_in[14];
  const float* rb1 = (const float*)d_in[15];
  const float* rW2 = (const float*)d_in[16];
  const float* rb2 = (const float*)d_in[17];
  _Float16* ws = (_Float16*)d_ws;   // needs >= 103,424 bytes of scratch

  prep_weights<<<(WS_HALVES + 255) / 256, 256, 0, stream>>>(dW1, dW2, dW3, gW1, gW2,
                                                            rW1, rW2, ws);
  nsde_kernel<<<kB / 64, 128, LDS_BYTES, stream>>>(x_path, t_span, dWn, ws, db1, db2,
                                                   db3, gb1, gb2, gsc, rb1, rb2,
                                                   (float*)d_out);
}
